// AtomAttentionEncoder_51780125720799
// MI455X (gfx1250) — compile-verified, matrix operand order checked by reading, not run on hardware
//
#include <hip/hip_runtime.h>
#include <hip/hip_bf16.h>

typedef __attribute__((ext_vector_type(16))) __bf16 v16bf;
typedef __attribute__((ext_vector_type(8)))  float  v8f;

#define N_ATOM   8192
#define CA       128
#define CP       16
#define NBLK     256
#define QBLK     32
#define KBLK     128
#define NPTS     (NBLK*QBLK*KBLK)   /* 1048576 */
#define NC       (N_ATOM*CA)        /* 1048576 */
#define HEADS    4
#define DHEAD    32
#define NLAYER   3
#define CTOK     384
#define NTOK     2048

// ---------------------------------------------------------------------------
// WMMA fragment helpers (CDNA5 wave32 layouts, CDNA5 ISA 7.12.2)
// ---------------------------------------------------------------------------
__device__ __forceinline__ int frag_k(int lane, int v) {
  int half = (lane >> 4) & 1;
  return ((v >> 2) << 4) + half * 8 + ((v & 3) << 1);   // pairs (k, k+1)
}

__device__ __forceinline__ v8f wmma_bf16(v16bf a, v16bf b, v8f c) {
  return __builtin_amdgcn_wmma_f32_16x16x32_bf16(false, a, false, b, (short)0, c, false, false);
}

// A fragment (16x32) from row-major bf16; two 16B runs per lane -> b128 loads.
// Also used for B fragments when the weight is stored transposed (ncols x kdim):
// the B layout mirrors A with the column index in place of the row index.
__device__ __forceinline__ v16bf load_a_g(const __bf16* X, size_t row, int ld, int kbase, int lane) {
  v16bf a;
  const __bf16* p = X + row * (size_t)ld + kbase;
#pragma unroll
  for (int v = 0; v < 8; ++v) {
    int k = frag_k(lane, v);
    a[2*v]   = p[k];
    a[2*v+1] = p[k+1];
  }
  return a;
}

// B fragment (32x16) from row-major bf16 W (kdim x ncols) -- gather fallback
__device__ __forceinline__ v16bf load_b_g(const __bf16* W, int kbase, int ld, int col, int lane) {
  v16bf b;
#pragma unroll
  for (int v = 0; v < 8; ++v) {
    int k = kbase + frag_k(lane, v);
    b[2*v]   = W[(size_t)k * ld + col];
    b[2*v+1] = W[(size_t)(k + 1) * ld + col];
  }
  return b;
}

// B fragment for 16xNcolReal weight, K zero-padded 16->32, cols >= ncolReal zero
__device__ __forceinline__ v16bf load_b16_pad(const __bf16* W, int lane, int ncolReal) {
  v16bf b; int n = lane & 15;
#pragma unroll
  for (int v = 0; v < 8; ++v) {
    int k = frag_k(lane, v);
    bool ok = (k < 16) && (n < ncolReal);
    b[2*v]   = ok ? W[k * ncolReal + n]       : (__bf16)0.0f;
    b[2*v+1] = ok ? W[(k + 1) * ncolReal + n] : (__bf16)0.0f;
  }
  return b;
}

// A fragment from row-major bf16 with 16 real K channels, zero-padded to 32
__device__ __forceinline__ v16bf load_a_g16_pad(const __bf16* X, size_t row, int lane) {
  v16bf a;
  const __bf16* p = X + row * 16;
#pragma unroll
  for (int v = 0; v < 8; ++v) {
    int k = frag_k(lane, v);
    a[2*v]   = (k < 16) ? p[k]     : (__bf16)0.0f;
    a[2*v+1] = (k < 16) ? p[k + 1] : (__bf16)0.0f;
  }
  return a;
}

// A fragment from a 16x17 (padded) f32 LDS tile, relu applied, K padded 16->32
__device__ __forceinline__ v16bf load_a_lds16_relu(const float (*t)[17], int lane) {
  v16bf a; int r = lane & 15;
#pragma unroll
  for (int v = 0; v < 8; ++v) {
    int k = frag_k(lane, v);
    float x0 = (k < 16) ? t[r][k]     : 0.f;
    float x1 = (k < 16) ? t[r][k + 1] : 0.f;
    a[2*v]   = (__bf16)fmaxf(x0, 0.f);
    a[2*v+1] = (__bf16)fmaxf(x1, 0.f);
  }
  return a;
}

__device__ __forceinline__ float sigmoidf(float x) { return 1.f / (1.f + expf(-x)); }

// ---------------------------------------------------------------------------
// Strip GEMM (fast path): Y = X @ W, W stored TRANSPOSED (ncols x kdim) so both
// A and B fragments are two 16B runs/lane (b128 loads). One wave = 16 rows x
// 64 cols (4 accumulators, A reused 4x). K-loop fully unrolled via template ->
// 16 (KD=128) / 32 (KD=256) back-to-back v_wmma per wave.
// ---------------------------------------------------------------------------
template<int KD>
__global__ void gemm_strip_kernel(const __bf16* __restrict__ X, const __bf16* __restrict__ Wt,
                                  float* __restrict__ Y, int nrows, int ncols, int relu) {
  int lane = threadIdx.x & 31, wave = threadIdx.x >> 5;
  int nstrip = ncols >> 6;                       // ncols % 64 == 0
  int total  = (nrows >> 4) * nstrip;
  int sid = blockIdx.x * 8 + wave;
  if (sid >= total) return;                      // wave-uniform: EXEC stays full
  int rt = sid / nstrip, cs = sid % nstrip;
  int row0 = rt << 4, col0 = cs << 6;
  int nn = lane & 15, half = (lane >> 4) & 1;

  v8f acc0 = {}, acc1 = {}, acc2 = {}, acc3 = {};
#pragma unroll
  for (int kb = 0; kb < KD; kb += 32) {
    v16bf a  = load_a_g(X,  (size_t)(row0 + nn),        KD, kb, lane);
    v16bf b0 = load_a_g(Wt, (size_t)(col0 +  0 + nn),   KD, kb, lane);
    v16bf b1 = load_a_g(Wt, (size_t)(col0 + 16 + nn),   KD, kb, lane);
    v16bf b2 = load_a_g(Wt, (size_t)(col0 + 32 + nn),   KD, kb, lane);
    v16bf b3 = load_a_g(Wt, (size_t)(col0 + 48 + nn),   KD, kb, lane);
    acc0 = wmma_bf16(a, b0, acc0);
    acc1 = wmma_bf16(a, b1, acc1);
    acc2 = wmma_bf16(a, b2, acc2);
    acc3 = wmma_bf16(a, b3, acc3);
  }
#pragma unroll
  for (int u = 0; u < 8; ++u) {
    int m = u + 8 * half;
    float y0 = acc0[u], y1 = acc1[u], y2 = acc2[u], y3 = acc3[u];
    if (relu) { y0 = fmaxf(y0, 0.f); y1 = fmaxf(y1, 0.f); y2 = fmaxf(y2, 0.f); y3 = fmaxf(y3, 0.f); }
    float* yr = Y + (size_t)(row0 + m) * ncols + col0;
    yr[nn]      = y0;
    yr[16 + nn] = y1;
    yr[32 + nn] = y2;
    yr[48 + nn] = y3;
  }
}

// Generic single-tile GEMM (used for ncols == 16 projections), W transposed too.
__global__ void gemm_tile_kernel(const __bf16* __restrict__ X, const __bf16* __restrict__ Wt,
                                 float* __restrict__ Y, int nrows, int ncols, int kdim, int relu) {
  int lane = threadIdx.x & 31, wave = threadIdx.x >> 5;
  int nct = ncols >> 4;
  int total = (nrows >> 4) * nct;
  int tile = blockIdx.x * 8 + wave;
  if (tile >= total) return;
  int rt = tile / nct, ct = tile % nct;
  int row0 = rt << 4, col0 = ct << 4;
  int nn = lane & 15, half = (lane >> 4) & 1;
  v8f acc = {};
  for (int kb = 0; kb < kdim; kb += 32) {
    v16bf a = load_a_g(X,  (size_t)(row0 + nn), kdim, kb, lane);
    v16bf b = load_a_g(Wt, (size_t)(col0 + nn), kdim, kb, lane);
    acc = wmma_bf16(a, b, acc);
  }
#pragma unroll
  for (int u = 0; u < 8; ++u) {
    int m = u + 8 * half;
    float v = acc[u];
    if (relu) v = fmaxf(v, 0.f);
    Y[(size_t)(row0 + m) * ncols + col0 + nn] = v;
  }
}

// ---------------------------------------------------------------------------
// Embedding + single_cond + LN(single_cond) (one block = one atom row, 128 thr)
// ---------------------------------------------------------------------------
__global__ void embed_ln_kernel(const float* __restrict__ pos, const float* __restrict__ mask,
                                const int* __restrict__ elem, const float* __restrict__ charge,
                                const int* __restrict__ name,
                                const float* __restrict__ w_pos, const float* __restrict__ w_mask,
                                const float* __restrict__ w_elem, const float* __restrict__ w_charge,
                                const float* __restrict__ w_name,
                                float* __restrict__ sc_out, float* __restrict__ a_out,
                                float* __restrict__ s_ln, __bf16* __restrict__ s_ln_b,
                                __bf16* __restrict__ relu_b) {
  int n = blockIdx.x, c = threadIdx.x;
  float p0 = pos[n*3+0], p1 = pos[n*3+1], p2 = pos[n*3+2];
  float mk = mask[n];
  float ch = charge[n];
  ch = logf(ch + sqrtf(ch*ch + 1.0f));     // arcsinh
  int e = elem[n];
  float act = p0*w_pos[c] + p1*w_pos[CA+c] + p2*w_pos[2*CA+c]
            + mk*w_mask[c] + w_elem[e*CA + c] + ch*w_charge[c];
#pragma unroll
  for (int i = 0; i < 4; ++i) {
    int nm = name[n*4 + i];
    act += w_name[(i*64 + nm)*CA + c];
  }
  float sc = act * mk;
  size_t idx = (size_t)n * CA + c;
  sc_out[idx] = sc;
  a_out[idx]  = sc;
  relu_b[idx] = (__bf16)fmaxf(sc, 0.f);

  __shared__ float buf[CA];
  buf[c] = sc; __syncthreads();
  for (int s = 64; s > 0; s >>= 1) { if (c < s) buf[c] += buf[c + s]; __syncthreads(); }
  float mean = buf[0] * (1.0f/128.0f);
  __syncthreads();
  float d = sc - mean;
  buf[c] = d * d; __syncthreads();
  for (int s = 64; s > 0; s >>= 1) { if (c < s) buf[c] += buf[c + s]; __syncthreads(); }
  float var = buf[0] * (1.0f/128.0f);
  float y = d * rsqrtf(var + 1e-5f);
  s_ln[idx] = y;
  s_ln_b[idx] = (__bf16)y;
}

// Row LayerNorm over 128 channels (f32 -> f32)
__global__ void ln128_kernel(const float* __restrict__ x, float* __restrict__ y) {
  int n = blockIdx.x, c = threadIdx.x;
  size_t idx = (size_t)n * CA + c;
  float v = x[idx];
  __shared__ float buf[CA];
  buf[c] = v; __syncthreads();
  for (int s = 64; s > 0; s >>= 1) { if (c < s) buf[c] += buf[c + s]; __syncthreads(); }
  float mean = buf[0] * (1.0f/128.0f);
  __syncthreads();
  float d = v - mean;
  buf[c] = d * d; __syncthreads();
  for (int s = 64; s > 0; s >>= 1) { if (c < s) buf[c] += buf[c + s]; __syncthreads(); }
  float var = buf[0] * (1.0f/128.0f);
  y[idx] = d * rsqrtf(var + 1e-5f);
}

// ---------------------------------------------------------------------------
// Pair init + 3x(relu -> 16x16 WMMA) MLP + residual + LN(16) -> bf16
// one wave = 16 points (consecutive keys of one query row)
// ---------------------------------------------------------------------------
__global__ void pair_kernel(const float* __restrict__ pos, const int* __restrict__ uid,
                            const float* __restrict__ rowv, const float* __restrict__ colv,
                            const float* __restrict__ w_off, const float* __restrict__ w_dist,
                            const float* __restrict__ w_pmask, const __bf16* __restrict__ wmlp_b,
                            float* __restrict__ pair_out, __bf16* __restrict__ pair_ln_b) {
  __shared__ float t1[8][16][17];
  __shared__ float t2[8][16][17];
  int lane = threadIdx.x & 31, wave = threadIdx.x >> 5;
  size_t pt0 = ((size_t)blockIdx.x * 8 + wave) * 16;
  int qrow = (int)(pt0 >> 7);
  int blk  = qrow >> 5;
  int kk0  = (int)(pt0 & 127);
  int start = blk * 32 - 48;
  start = start < 0 ? 0 : (start > (N_ATOM - KBLK) ? (N_ATOM - KBLK) : start);

  int p   = lane & 15;
  int chb = ((lane >> 4) & 1) * 8;
  int i = qrow;
  int j = start + kk0 + p;
  float ox = pos[i*3+0] - pos[j*3+0];
  float oy = pos[i*3+1] - pos[j*3+1];
  float oz = pos[i*3+2] - pos[j*3+2];
  float invd = 1.f / (1.f + ox*ox + oy*oy + oz*oz);
  float valid = (uid[i] == uid[(j >> 2) << 2]) ? 1.f : 0.f;

  float pv[8];
#pragma unroll
  for (int u = 0; u < 8; ++u) {
    int c = chb + u;
    float v = valid * (ox*w_off[c] + oy*w_off[16+c] + oz*w_off[32+c]
                       + invd*w_dist[c] + w_pmask[c]);
    v += rowv[(size_t)i*16 + c] + colv[(size_t)j*16 + c];
    pv[u] = v;
    t1[wave][p][c] = v;
  }
  __builtin_amdgcn_wave_barrier();

  int half = (lane >> 4) & 1, nn = lane & 15;
  // h1 = relu(pair) @ w_mlp1
  {
    v16bf a = load_a_lds16_relu(t1[wave], lane);
    v16bf b = load_b16_pad(wmlp_b + 0*256, lane, 16);
    v8f acc = {}; acc = wmma_bf16(a, b, acc);
#pragma unroll
    for (int u = 0; u < 8; ++u) t2[wave][u + 8*half][nn] = acc[u];
  }
  __builtin_amdgcn_wave_barrier();
  // h2 = relu(h1) @ w_mlp2
  {
    v16bf a = load_a_lds16_relu(t2[wave], lane);
    v16bf b = load_b16_pad(wmlp_b + 1*256, lane, 16);
    v8f acc = {}; acc = wmma_bf16(a, b, acc);
#pragma unroll
    for (int u = 0; u < 8; ++u) t1[wave][u + 8*half][nn] = acc[u];
  }
  __builtin_amdgcn_wave_barrier();
  // h3 = relu(h2) @ w_mlp3
  {
    v16bf a = load_a_lds16_relu(t1[wave], lane);
    v16bf b = load_b16_pad(wmlp_b + 2*256, lane, 16);
    v8f acc = {}; acc = wmma_bf16(a, b, acc);
#pragma unroll
    for (int u = 0; u < 8; ++u) t2[wave][u + 8*half][nn] = acc[u];
  }
  __builtin_amdgcn_wave_barrier();

  // pair = pair + h3 ; store f32 and LDS
#pragma unroll
  for (int u = 0; u < 8; ++u) {
    int c = chb + u;
    float v = pv[u] + t2[wave][p][c];
    t1[wave][p][c] = v;
    pair_out[pt0 * 16 + (size_t)p * 16 + c] = v;
  }
  __builtin_amdgcn_wave_barrier();

  // LN over 16 channels (lanes 0..15 each own one point)
  if (lane < 16) {
    float s = 0.f;
#pragma unroll
    for (int c = 0; c < 16; ++c) s += t1[wave][lane][c];
    float mean = s * (1.f/16.f);
    float vs = 0.f;
#pragma unroll
    for (int c = 0; c < 16; ++c) { float d = t1[wave][lane][c] - mean; vs += d * d; }
    float inv = rsqrtf(vs * (1.f/16.f) + 1e-5f);
#pragma unroll
    for (int c = 0; c < 16; ++c)
      pair_ln_b[(pt0 + lane) * 16 + c] = (__bf16)((t1[wave][lane][c] - mean) * inv);
  }
}

// pair bias per layer: pbias[point,h] = pair_ln[point,:16] @ w_pb[l][:, h]  (WMMA, K/N padded)
__global__ void pairbias_kernel(const __bf16* __restrict__ pln, const __bf16* __restrict__ wpb,
                                float* __restrict__ pbias) {
  int lane = threadIdx.x & 31, wave = threadIdx.x >> 5;
  size_t pt0 = ((size_t)blockIdx.x * 8 + wave) * 16;
  v16bf a = load_a_g16_pad(pln, pt0 + (lane & 15), lane);
  v16bf b = load_b16_pad(wpb, lane, 4);
  v8f acc = {}; acc = wmma_bf16(a, b, acc);
  int half = (lane >> 4) & 1, nn = lane & 15;
  if (nn < 4) {
#pragma unroll
    for (int u = 0; u < 8; ++u)
      pbias[(pt0 + u + 8 * half) * 4 + nn] = acc[u];
  }
}

// ---------------------------------------------------------------------------
// Block-local attention: one workgroup per block b; wave = (head, q-half)
// ---------------------------------------------------------------------------
__global__ void attn_kernel(const __bf16* __restrict__ qb, const __bf16* __restrict__ kb,
                            const __bf16* __restrict__ vb, const float* __restrict__ pbias,
                            float* __restrict__ o) {
  __shared__ float lg[8][16][128];         // 64KB logits/probs
  int lane = threadIdx.x & 31, wave = threadIdx.x >> 5;
  int b = blockIdx.x;
  int h = wave >> 1, qh = wave & 1;
  int start = b * 32 - 48;
  start = start < 0 ? 0 : (start > (N_ATOM - KBLK) ? (N_ATOM - KBLK) : start);
  int i0 = b * 32 + qh * 16;
  int half = (lane >> 4) & 1, nn = lane & 15;
  const float scale = 0.17677669529663687f;   // 1/sqrt(32)

  // Prefetch V rows (consumed after softmax) while we do QK^T + softmax.
  // Each lane prefetches a distinct key row -> every 4th of the 128 keys of this block.
  __builtin_prefetch(vb + (size_t)(start + lane * 4) * CA + h * DHEAD, 0, 0);

  v16bf aq = load_a_g(qb, (size_t)(i0 + nn), CA, h * DHEAD, lane);
#pragma unroll
  for (int t = 0; t < 8; ++t) {
    v16bf bk;
    int key = start + t * 16 + nn;
#pragma unroll
    for (int v = 0; v < 8; ++v) {
      int k = frag_k(lane, v);
      bk[2*v]   = kb[(size_t)key * CA + h * DHEAD + k];
      bk[2*v+1] = kb[(size_t)key * CA + h * DHEAD + k + 1];
    }
    v8f acc = {}; acc = wmma_bf16(aq, bk, acc);
#pragma unroll
    for (int u = 0; u < 8; ++u) {
      int m = u + 8 * half;
      size_t pt = (size_t)(b * 32 + qh * 16 + m) * 128 + t * 16 + nn;
      lg[wave][m][t * 16 + nn] = acc[u] * scale + pbias[pt * 4 + h];
    }
  }
  __syncthreads();

  if (lane < 16) {
    int r = lane;
    float mx = -1e30f;
    for (int k2 = 0; k2 < 128; ++k2) mx = fmaxf(mx, lg[wave][r][k2]);
    float s = 0.f;
    for (int k2 = 0; k2 < 128; ++k2) { float e = expf(lg[wave][r][k2] - mx); lg[wave][r][k2] = e; s += e; }
    float inv = 1.f / s;
    for (int k2 = 0; k2 < 128; ++k2) lg[wave][r][k2] *= inv;
  }
  __syncthreads();

  v16bf ap[4];
#pragma unroll
  for (int ks = 0; ks < 4; ++ks) {
#pragma unroll
    for (int v = 0; v < 8; ++v) {
      int k = ks * 32 + frag_k(lane, v);
      ap[ks][2*v]   = (__bf16)lg[wave][lane & 15][k];
      ap[ks][2*v+1] = (__bf16)lg[wave][lane & 15][k + 1];
    }
  }
#pragma unroll
  for (int nt = 0; nt < 2; ++nt) {
    v8f acc = {};
#pragma unroll
    for (int ks = 0; ks < 4; ++ks) {
      v16bf bv;
#pragma unroll
      for (int v = 0; v < 8; ++v) {
        int k = frag_k(lane, v);
        bv[2*v]   = vb[(size_t)(start + ks * 32 + k)     * CA + h * DHEAD + nt * 16 + nn];
        bv[2*v+1] = vb[(size_t)(start + ks * 32 + k + 1) * CA + h * DHEAD + nt * 16 + nn];
      }
      acc = wmma_bf16(ap[ks], bv, acc);
    }
#pragma unroll
    for (int u = 0; u < 8; ++u) {
      int m = u + 8 * half;
      o[(size_t)(i0 + m) * CA + h * DHEAD + nt * 16 + nn] = acc[u];
    }
  }
}

// ---------------------------------------------------------------------------
// Elementwise kernels
// ---------------------------------------------------------------------------
__global__ void cvt_f32_bf16_kernel(const float* __restrict__ x, __bf16* __restrict__ y, int n) {
  int i = blockIdx.x * blockDim.x + threadIdx.x;
  if (i < n) y[i] = (__bf16)x[i];
}
// f32 (nmat x kdim x ncols) -> bf16 TRANSPOSED (nmat x ncols x kdim)
__global__ void cvtT_kernel(const float* __restrict__ W, __bf16* __restrict__ Wt,
                            int kdim, int ncols, int nmat) {
  int i = blockIdx.x * blockDim.x + threadIdx.x;
  int per = kdim * ncols;
  if (i < per * nmat) {
    int m = i / per, r = i - m * per;
    int k = r / ncols, c = r - k * ncols;
    Wt[(size_t)m * per + (size_t)c * kdim + k] = (__bf16)W[i];
  }
}
__global__ void biascvt_kernel(const float* __restrict__ x, const float* __restrict__ bias,
                               __bf16* __restrict__ y, int n) {
  int i = blockIdx.x * blockDim.x + threadIdx.x;
  if (i < n) y[i] = (__bf16)(x[i] + bias[i & 127]);
}
__global__ void gate_kernel(const float* __restrict__ g, const float* __restrict__ bias,
                            const float* __restrict__ aln, const float* __restrict__ b,
                            __bf16* __restrict__ out, int n) {
  int i = blockIdx.x * blockDim.x + threadIdx.x;
  if (i < n) out[i] = (__bf16)(sigmoidf(g[i] + bias[i & 127]) * aln[i] + b[i]);
}
__global__ void go_kernel(const float* __restrict__ g, const float* __restrict__ o,
                          __bf16* __restrict__ out, int n) {
  int i = blockIdx.x * blockDim.x + threadIdx.x;
  if (i < n) out[i] = (__bf16)(sigmoidf(g[i]) * o[i]);
}
__global__ void hid_kernel(const float* __restrict__ h1, const float* __restrict__ h2,
                           __bf16* __restrict__ out, int n) {
  int i = blockIdx.x * blockDim.x + threadIdx.x;
  if (i < n) { float x = h1[i]; out[i] = (__bf16)(x * sigmoidf(x) * h2[i]); }
}
__global__ void anew_kernel(const float* __restrict__ sg, const float* __restrict__ o2,
                            const float* __restrict__ sgt, const float* __restrict__ t,
                            float* __restrict__ a, int n) {
  int i = blockIdx.x * blockDim.x + threadIdx.x;
  if (i < n) a[i] = sigmoidf(sg[i] - 2.f) * o2[i] + sigmoidf(sgt[i] - 2.f) * t[i];
}
__global__ void token_kernel(const float* __restrict__ tokr, const float* __restrict__ mask,
                             float* __restrict__ out) {
  int t = blockIdx.x, c = threadIdx.x;   // 384 threads
  float s = 0.f, sm = 0.f;
#pragma unroll
  for (int a = 0; a < 4; ++a) {
    float m = mask[t * 4 + a];
    s += tokr[(size_t)(t * 4 + a) * CTOK + c] * m;
    sm += m;
  }
  out[(size_t)t * CTOK + c] = s / fmaxf(sm, 1e-6f);
}

// ---------------------------------------------------------------------------
// Host launcher
// ---------------------------------------------------------------------------
extern "C" void kernel_launch(void* const* d_in, const int* in_sizes, int n_in,
                              void* d_out, int out_size, void* d_ws, size_t ws_size,
                              hipStream_t stream) {
  const float* positions = (const float*)d_in[0];
  const float* mask      = (const float*)d_in[1];
  const int*   element   = (const int*)d_in[2];
  const float* charge    = (const float*)d_in[3];
  const int*   name      = (const int*)d_in[4];
  const int*   uid       = (const int*)d_in[5];
  int pi = 6;
  const float* w_ref_pos    = (const float*)d_in[pi++];
  const float* w_ref_mask   = (const float*)d_in[pi++];
  const float* w_ref_elem   = (const float*)d_in[pi++];
  const float* w_ref_charge = (const float*)d_in[pi++];
  const float* w_ref_name   = (const float*)d_in[pi++];
  const float* w_row   = (const float*)d_in[pi++];
  const float* w_col   = (const float*)d_in[pi++];
  const float* w_off   = (const float*)d_in[pi++];
  const float* w_dist  = (const float*)d_in[pi++];
  const float* w_pmask = (const float*)d_in[pi++];
  const float* w_mlp1  = (const float*)d_in[pi++];
  const float* w_mlp2  = (const float*)d_in[pi++];
  const float* w_mlp3  = (const float*)d_in[pi++];
  const float* wg_q    = (const float*)d_in[pi++];
  const float* bg_q    = (const float*)d_in[pi++];
  const float* wb_q    = (const float*)d_in[pi++];
  const float* wg_kv   = (const float*)d_in[pi++];
  const float* bg_kv   = (const float*)d_in[pi++];
  const float* wb_kv   = (const float*)d_in[pi++];
  const float* wq      = (const float*)d_in[pi++];
  const float* bq      = (const float*)d_in[pi++];
  const float* wk      = (const float*)d_in[pi++];
  const float* wv      = (const float*)d_in[pi++];
  const float* w_pb    = (const float*)d_in[pi++];
  const float* wg_attn = (const float*)d_in[pi++];
  const float* wo      = (const float*)d_in[pi++];
  const float* ws_gate = (const float*)d_in[pi++];
  const float* wg_t    = (const float*)d_in[pi++];
  const float* bg_t    = (const float*)d_in[pi++];
  const float* wb_t    = (const float*)d_in[pi++];
  const float* w_t1    = (const float*)d_in[pi++];
  const float* w_t2    = (const float*)d_in[pi++];
  const float* w_t3    = (const float*)d_in[pi++];
  const float* ws_gate_t = (const float*)d_in[pi++];
  const float* w_proj  = (const float*)d_in[pi++];

  float* out_tok  = (float*)d_out;
  float* out_a    = out_tok + (size_t)NTOK * CTOK;
  float* out_sc   = out_a + (size_t)NC;
  float* out_pair = out_sc + (size_t)NC;

  // bump allocator over d_ws
  char* wsb = (char*)d_ws; size_t off = 0;
  auto AF = [&](size_t n) { float* p = (float*)(wsb + off); off += ((n * 4 + 255) & ~(size_t)255); return p; };
  auto AB = [&](size_t n) { __bf16* p = (__bf16*)(wsb + off); off += ((n * 2 + 255) & ~(size_t)255); return p; };

  float *s_ln = AF(NC), *rowv = AF(N_ATOM*16), *colv = AF(N_ATOM*16);
  float *abuf = AF(NC), *a_ln = AF(NC), *gbuf = AF(NC), *bbuf = AF(NC);
  float *qf = AF(NC), *kf = AF(NC), *vf = AF(NC);
  float *gattn = AF(NC), *sg = AF(NC), *sgt = AF(NC);
  float *obuf = AF(NC), *o2 = AF(NC);
  float *h1 = AF((size_t)N_ATOM*256), *h2 = AF((size_t)N_ATOM*256), *tbuf = AF(NC);
  float *pbias = AF((size_t)NPTS*4), *tokr = AF((size_t)N_ATOM*CTOK);
  __bf16 *s_ln_b = AB(NC), *relu_b = AB(NC), *pair_ln_b = AB((size_t)NPTS*16);
  __bf16 *aq_b = AB(NC), *akv_b = AB(NC), *at_b = AB(NC);
  __bf16 *q_b = AB(NC), *k_b = AB(NC), *v_b = AB(NC);
  __bf16 *go_b = AB(NC), *hid_b = AB((size_t)N_ATOM*256), *a_b = AB(NC);
  // transposed bf16 weights (ncols x kdim per matrix)
  __bf16 *w_row_b = AB(CA*16), *w_col_b = AB(CA*16), *wmlp_b = AB(3*256), *wpb_b = AB(3*64);
  __bf16 *wgq_b = AB(3*CA*CA), *wbq_b = AB(3*CA*CA), *wgkv_b = AB(3*CA*CA), *wbkv_b = AB(3*CA*CA);
  __bf16 *wgt_b = AB(3*CA*CA), *wbt_b = AB(3*CA*CA);
  __bf16 *wq_b = AB(3*CA*CA), *wk_b = AB(3*CA*CA), *wv_b = AB(3*CA*CA);
  __bf16 *wgat_b = AB(3*CA*CA), *wo_b = AB(3*CA*CA), *wsg_b = AB(3*CA*CA), *wsgt_b = AB(3*CA*CA);
  __bf16 *wt1_b = AB(3*CA*256), *wt2_b = AB(3*CA*256), *wt3_b = AB(3*256*CA), *wproj_b = AB(CA*CTOK);
  (void)ws_size; (void)in_sizes; (void)n_in; (void)out_size;

  auto CVT = [&](const float* src, __bf16* dst, int n) {
    cvt_f32_bf16_kernel<<<dim3((n + 255) / 256), dim3(256), 0, stream>>>(src, dst, n);
  };
  auto CVTT = [&](const float* src, __bf16* dst, int kd, int nc, int nm) {
    int n = kd * nc * nm;
    cvtT_kernel<<<dim3((n + 255) / 256), dim3(256), 0, stream>>>(src, dst, kd, nc, nm);
  };
  auto GEMM = [&](const __bf16* X, const __bf16* Wt, float* Y, int nr, int nc, int kd, int relu) {
    if ((nc & 63) == 0 && kd == 128) {
      int strips = (nr / 16) * (nc / 64);
      gemm_strip_kernel<128><<<dim3((strips + 7) / 8), dim3(256), 0, stream>>>(X, Wt, Y, nr, nc, relu);
    } else if ((nc & 63) == 0 && kd == 256) {
      int strips = (nr / 16) * (nc / 64);
      gemm_strip_kernel<256><<<dim3((strips + 7) / 8), dim3(256), 0, stream>>>(X, Wt, Y, nr, nc, relu);
    } else {
      int tiles = (nr / 16) * (nc / 16);
      gemm_tile_kernel<<<dim3((tiles + 7) / 8), dim3(256), 0, stream>>>(X, Wt, Y, nr, nc, kd, relu);
    }
  };

  // --- weight conversions (f32 -> bf16, transposed for the GEMM B path) ---
  CVTT(w_row, w_row_b, CA, 16, 1);  CVTT(w_col, w_col_b, CA, 16, 1);
  CVT(w_mlp1, wmlp_b + 0*256, 256); CVT(w_mlp2, wmlp_b + 1*256, 256); CVT(w_mlp3, wmlp_b + 2*256, 256);
  CVT(w_pb, wpb_b, 3*64);
  CVTT(wg_q, wgq_b, CA, CA, 3);   CVTT(wb_q, wbq_b, CA, CA, 3);
  CVTT(wg_kv, wgkv_b, CA, CA, 3); CVTT(wb_kv, wbkv_b, CA, CA, 3);
  CVTT(wg_t, wgt_b, CA, CA, 3);   CVTT(wb_t, wbt_b, CA, CA, 3);
  CVTT(wq, wq_b, CA, CA, 3);      CVTT(wk, wk_b, CA, CA, 3);     CVTT(wv, wv_b, CA, CA, 3);
  CVTT(wg_attn, wgat_b, CA, CA, 3); CVTT(wo, wo_b, CA, CA, 3);
  CVTT(ws_gate, wsg_b, CA, CA, 3);  CVTT(ws_gate_t, wsgt_b, CA, CA, 3);
  CVTT(w_t1, wt1_b, CA, 256, 3);  CVTT(w_t2, wt2_b, CA, 256, 3); CVTT(w_t3, wt3_b, 256, CA, 3);
  CVTT(w_proj, wproj_b, CA, CTOK, 1);

  // --- embedding + LN(single_cond) ---
  embed_ln_kernel<<<dim3(N_ATOM), dim3(CA), 0, stream>>>(
      positions, mask, element, charge, name,
      w_ref_pos, w_ref_mask, w_ref_elem, w_ref_charge, w_ref_name,
      out_sc, abuf, s_ln, s_ln_b, relu_b);

  // --- row / col projections (WMMA) ---
  GEMM(relu_b, w_row_b, rowv, N_ATOM, 16, CA, 0);
  GEMM(relu_b, w_col_b, colv, N_ATOM, 16, CA, 0);

  // --- pair tensor + MLP + LN (WMMA) ---
  pair_kernel<<<dim3(NPTS / 128), dim3(256), 0, stream>>>(
      positions, uid, rowv, colv, w_off, w_dist, w_pmask, wmlp_b, out_pair, pair_ln_b);

  const int ncElt = NC, nc2Elt = N_ATOM * 256;
  dim3 eb(256), egNC((ncElt + 255) / 256), eg2((nc2Elt + 255) / 256);

  for (int l = 0; l < NLAYER; ++l) {
    size_t wofs = (size_t)l * CA * CA, bofs = (size_t)l * CA, w2 = (size_t)l * CA * 256;
    // a_ln = LN(a)
    ln128_kernel<<<dim3(N_ATOM), dim3(CA), 0, stream>>>(abuf, a_ln);
    // aq / akv / at (gated linear units)
    GEMM(s_ln_b, wgq_b + wofs, gbuf, N_ATOM, CA, CA, 0);
    GEMM(s_ln_b, wbq_b + wofs, bbuf, N_ATOM, CA, CA, 0);
    gate_kernel<<<egNC, eb, 0, stream>>>(gbuf, bg_q + bofs, a_ln, bbuf, aq_b, ncElt);
    GEMM(s_ln_b, wgkv_b + wofs, gbuf, N_ATOM, CA, CA, 0);
    GEMM(s_ln_b, wbkv_b + wofs, bbuf, N_ATOM, CA, CA, 0);
    gate_kernel<<<egNC, eb, 0, stream>>>(gbuf, bg_kv + bofs, a_ln, bbuf, akv_b, ncElt);
    GEMM(s_ln_b, wgt_b + wofs, gbuf, N_ATOM, CA, CA, 0);
    GEMM(s_ln_b, wbt_b + wofs, bbuf, N_ATOM, CA, CA, 0);
    gate_kernel<<<egNC, eb, 0, stream>>>(gbuf, bg_t + bofs, a_ln, bbuf, at_b, ncElt);
    // q, k, v
    GEMM(aq_b,  wq_b + wofs, qf, N_ATOM, CA, CA, 0);
    GEMM(akv_b, wk_b + wofs, kf, N_ATOM, CA, CA, 0);
    GEMM(akv_b, wv_b + wofs, vf, N_ATOM, CA, CA, 0);
    biascvt_kernel<<<egNC, eb, 0, stream>>>(qf, bq + bofs, q_b, ncElt);
    cvt_f32_bf16_kernel<<<egNC, eb, 0, stream>>>(kf, k_b, ncElt);
    cvt_f32_bf16_kernel<<<egNC, eb, 0, stream>>>(vf, v_b, ncElt);
    // attention gate + pair bias + attention
    GEMM(aq_b, wgat_b + wofs, gattn, N_ATOM, CA, CA, 0);
    pairbias_kernel<<<dim3(NPTS / 128), dim3(256), 0, stream>>>(pair_ln_b, wpb_b + l * 64, pbias);
    attn_kernel<<<dim3(NBLK), dim3(256), 0, stream>>>(q_b, k_b, v_b, pbias, obuf);
    // o2 = (sigmoid(g) * o) @ wo ; sg
    go_kernel<<<egNC, eb, 0, stream>>>(gattn, obuf, go_b, ncElt);
    GEMM(go_b, wo_b + wofs, o2, N_ATOM, CA, CA, 0);
    GEMM(s_ln_b, wsg_b + wofs, sg, N_ATOM, CA, CA, 0);
    // transition
    GEMM(at_b, wt1_b + w2, h1, N_ATOM, 256, CA, 0);
    GEMM(at_b, wt2_b + w2, h2, N_ATOM, 256, CA, 0);
    hid_kernel<<<eg2, eb, 0, stream>>>(h1, h2, hid_b, nc2Elt);
    GEMM(hid_b, wt3_b + w2, tbuf, N_ATOM, CA, 256, 0);
    GEMM(s_ln_b, wsgt_b + wofs, sgt, N_ATOM, CA, CA, 0);
    // a = sigmoid(sg-2)*o2 + sigmoid(sgt-2)*t
    anew_kernel<<<egNC, eb, 0, stream>>>(sg, o2, sgt, tbuf, abuf, ncElt);
  }

  // final a -> d_out ; token projection
  hipMemcpyAsync(out_a, abuf, (size_t)NC * 4, hipMemcpyDeviceToDevice, stream);
  cvt_f32_bf16_kernel<<<egNC, eb, 0, stream>>>(abuf, a_b, ncElt);
  GEMM(a_b, wproj_b, tokr, N_ATOM, CTOK, CA, 1);
  token_kernel<<<dim3(NTOK), dim3(CTOK), 0, stream>>>(tokr, mask, out_tok);
}
